// FunctionClassifier_75608604278842
// MI455X (gfx1250) — matrix operand order repaired and checked by reading, not compiled, not run on hardware
//
#include <hip/hip_runtime.h>
#include <hip/hip_bf16.h>

typedef __attribute__((ext_vector_type(16))) __bf16 v16bf;
typedef __attribute__((ext_vector_type(8)))  __bf16 v8bf;
typedef __attribute__((ext_vector_type(8)))  float  v8f;

#define T_STEPS 2048
#define BATCH   64
#define HID     256

// ---- workspace layout (bytes) ----
#define OFF_FLAGS 0                       // 3 ints (layer progress)
#define OFF_HLAST 4096                    // 64*256 f32 = 64KB
#define OFF_WGT   131072                  // 3 * [512 n][512 k] bf16 = 1.5MB
#define OFF_WCT   2097152                 // 3 * [256 n][512 k] bf16 = 0.75MB
#define OFF_YS0   4194304                 // 2048*64*256 bf16 = 64MB
#define OFF_YS1   (4194304 + 67108864)    // 64MB

// ---------------------------------------------------------------------------
// Prep: fp32 weights -> bf16, transposed to [N][K] (each lane's 16 B-values
// contiguous 32B), layer-0 K=257 zero-padded to K=512 (k=0 -> x row,
// k in [1,255] -> 0, k in [256,511] -> h rows). Also zero progress flags.
// ---------------------------------------------------------------------------
__global__ void prep_weights(const float* __restrict__ Wg0, const float* __restrict__ Wc0,
                             const float* __restrict__ Wg1, const float* __restrict__ Wc1,
                             const float* __restrict__ Wg2, const float* __restrict__ Wc2,
                             char* __restrict__ ws) {
  __bf16* WgT  = (__bf16*)(ws + OFF_WGT);
  __bf16* WcT  = (__bf16*)(ws + OFF_WCT);
  int*    flags = (int*)(ws + OFF_FLAGS);
  long i = (long)blockIdx.x * blockDim.x + threadIdx.x;
  if (i < 4) flags[i] = 0;
  const long NG = 3L * 512 * 512;
  const long NC = 3L * 256 * 512;
  if (i < NG) {
    int layer = (int)(i >> 18);
    int rem   = (int)(i & 262143);
    int n = rem >> 9, k = rem & 511;
    const float* W = (layer == 0) ? Wg0 : (layer == 1) ? Wg1 : Wg2;
    float v;
    if (layer == 0) {
      if (k == 0)        v = W[n];
      else if (k < 256)  v = 0.0f;
      else               v = W[(k - 255) * 512 + n];
    } else {
      v = W[k * 512 + n];
    }
    WgT[i] = (__bf16)v;
  } else if (i < NG + NC) {
    long j = i - NG;
    int layer = (int)(j >> 17);
    int rem   = (int)(j & 131071);
    int n = rem >> 9, k = rem & 511;
    const float* W = (layer == 0) ? Wc0 : (layer == 1) ? Wc1 : Wc2;
    float v;
    if (layer == 0) {
      if (k == 0)        v = W[n];
      else if (k < 256)  v = 0.0f;
      else               v = W[(k - 255) * 256 + n];
    } else {
      v = W[k * 256 + n];
    }
    WcT[j] = (__bf16)v;
  }
}

// ---------------------------------------------------------------------------
// Fragment loaders (bases pre-include row/column + lane-octet offsets, so all
// K-step addressing constant-folds to immediate offsets).
// ---------------------------------------------------------------------------
__device__ __forceinline__ v16bf lda_at(const __bf16* Abase, int kcol) {
  v8bf lo = *(const v8bf*)(Abase + kcol);
  v8bf hi = *(const v8bf*)(Abase + kcol + 8);
  return __builtin_shufflevector(lo, hi, 0,1,2,3,4,5,6,7,8,9,10,11,12,13,14,15);
}
__device__ __forceinline__ v16bf ldb_at(const __bf16* Wbase, int ksw) {
  return *(const v16bf*)(Wbase + ksw * 32);
}

// Fully unrolled, distance-2 software-pipelined K chain.
// L0: layer-0 skips padded-zero K blocks 1..7 (9 steps instead of 16).
// CAND: candidate reads A cols [0,256) (x) then [512,768) (r*h).
template <bool CAND, bool L0>
__device__ __forceinline__ v8f gemm_job(const __bf16* __restrict__ Abase,
                                        const __bf16* __restrict__ Wbase) {
  constexpr int NKS = L0 ? 9 : 16;
  v16bf av[3], bv[3];
  v8f acc = {};
  #pragma unroll
  for (int p = 0; p < 2; ++p) {
    const int k = (L0 && p > 0) ? p + 7 : p;
    const int kcolA = CAND ? ((k < 8) ? k * 32 : 512 + (k - 8) * 32) : k * 32;
    av[p] = lda_at(Abase, kcolA);
    bv[p] = ldb_at(Wbase, k);
  }
  #pragma unroll
  for (int i = 0; i < NKS; ++i) {
    const int pf = i + 2;
    if (pf < NKS) {
      const int k = (L0 && pf > 0) ? pf + 7 : pf;
      const int kcolA = CAND ? ((k < 8) ? k * 32 : 512 + (k - 8) * 32) : k * 32;
      av[pf % 3] = lda_at(Abase, kcolA);
      bv[pf % 3] = ldb_at(Wbase, k);
    }
    __builtin_amdgcn_sched_barrier(0);   // keep prefetch(i+2) ahead of wmma(i)
    acc = __builtin_amdgcn_wmma_f32_16x16x32_bf16(false, av[i % 3], false, bv[i % 3],
                                                  (short)0, acc, false, false);
    __builtin_amdgcn_sched_barrier(0);
  }
  return acc;
}

__device__ __forceinline__ float sigmoid_f(float x) { return 1.0f / (1.0f + __expf(-x)); }
__device__ __forceinline__ float tanh_f(float x)    { return 1.0f - 2.0f / (__expf(2.0f * x) + 1.0f); }

// ---------------------------------------------------------------------------
// Persistent pipelined GRU stack: grid.x = 3 (block = layer on one WGP),
// 1024 threads = 32 wave32 waves. LDS: A[64][768] bf16 | H[64][256] f32 |
// U[64][256] f32. A cols: [0,256)=x, [256,512)=h_{t-1}, [512,768)=r*h.
// ---------------------------------------------------------------------------
__global__ void __launch_bounds__(1024, 1)
gru_stack(const float* __restrict__ x,
          const float* __restrict__ bg0, const float* __restrict__ bc0,
          const float* __restrict__ bg1, const float* __restrict__ bc1,
          const float* __restrict__ bg2, const float* __restrict__ bc2,
          char* __restrict__ ws) {
  extern __shared__ char smem[];
  __bf16* A = (__bf16*)smem;                                   // 98304 B
  float*  H = (float*)(smem + 64 * 768 * 2);                   // 65536 B
  float*  U = (float*)(smem + 64 * 768 * 2 + 64 * 256 * 4);    // 65536 B

  const int L    = blockIdx.x;
  const int tid  = threadIdx.x;
  const int wave = tid >> 5;
  const int lane = tid & 31;

  const __bf16* WgT = (const __bf16*)(ws + OFF_WGT) + (long)L * 512 * 512;
  const __bf16* WcT = (const __bf16*)(ws + OFF_WCT) + (long)L * 256 * 512;
  const float*  bg  = (L == 0) ? bg0 : (L == 1) ? bg1 : bg2;
  const float*  bc  = (L == 0) ? bc0 : (L == 1) ? bc1 : bc2;
  __bf16*       ysOut = (__bf16*)(ws + ((L == 1) ? OFF_YS1 : OFF_YS0));
  const __bf16* ysIn  = (const __bf16*)(ws + ((L == 1) ? OFF_YS0 : OFF_YS1));
  int*   flags = (int*)(ws + OFF_FLAGS);
  float* hlast = (float*)(ws + OFF_HLAST);

  // ---- loop-invariant decode ----
  const bool l0  = (L == 0);
  const int  mt     = wave & 3;           // wave's M tile (same for all its jobs)
  const int  lane15 = lane & 15;
  const int  koctA  = (lane & 16) ? 16 : 0;
  const int  koctB  = (lane & 16) ? 16 : 0;
  const int  mrow0  = mt * 16 + ((lane & 16) ? 8 : 0);
  const __bf16* Abase = A + (mt * 16 + lane15) * 768 + koctA;

  const __bf16* WgBase[4];  float bgv[4];  int nG[4];
  #pragma unroll
  for (int j = 0; j < 4; ++j) {                  // gates N tiles: 0..31
    int nt = j * 8 + (wave >> 2);
    nG[j] = nt * 16 + lane15;
    WgBase[j] = WgT + (long)nG[j] * 512 + koctB;
    bgv[j] = bg[nG[j]];
  }
  const __bf16* WcBase[2];  float bcv[2];  int nC[2];
  #pragma unroll
  for (int j = 0; j < 2; ++j) {                  // candidate N tiles: 0..15
    int nt = j * 8 + (wave >> 2);
    nC[j] = nt * 16 + lane15;
    WcBase[j] = WcT + (long)nC[j] * 512 + koctB;
    bcv[j] = bc[nC[j]];
  }

  // init: zero A (covers layer-0 pad cols + h0 = 0) and H
  for (int i = tid; i < 64 * 768; i += 1024) A[i] = (__bf16)0.0f;
  for (int i = tid; i < 64 * 256; i += 1024) H[i] = 0.0f;
  __syncthreads();

  for (int t = 0; t < T_STEPS; ++t) {
    // ---- wait for previous layer's timestep t ----
    if (L > 0) {
      while (__hip_atomic_load(&flags[L - 1], __ATOMIC_ACQUIRE, __HIP_MEMORY_SCOPE_AGENT) < t + 1)
        __builtin_amdgcn_s_sleep(1);
    }
    // ---- stage input slice of A ----
    if (l0) {
      if (tid < 64) A[tid * 768] = (__bf16)x[tid * T_STEPS + t];
    } else {
      const __bf16* src = ysIn + (long)t * BATCH * HID;
      for (int c = tid; c < 2048; c += 1024) {        // 2048 x 16B chunks
        int b = c >> 5;
        int k = (c & 31) << 3;
        *(v8bf*)(A + b * 768 + k) = *(const v8bf*)(src + b * 256 + k);
      }
    }
    __syncthreads();

    // ---- gates: ru = sigmoid([x,h] @ Wg + bg) -> r*h, u into LDS ----
    #pragma unroll
    for (int j = 0; j < 4; ++j) {
      v8f acc = l0 ? gemm_job<false, true >(Abase, WgBase[j])
                   : gemm_job<false, false>(Abase, WgBase[j]);
      int n = nG[j];
      float bias = bgv[j];
      #pragma unroll
      for (int i = 0; i < 8; ++i) {
        int m = mrow0 + i;
        float v = sigmoid_f(acc[i] + bias);
        if (n < 256) {                                // r gate -> r*h in A[:,512+n]
          A[m * 768 + 512 + n] = (__bf16)(v * H[m * 256 + n]);
        } else {                                      // u gate
          U[m * 256 + (n - 256)] = v;
        }
      }
    }
    __syncthreads();

    // ---- candidate + state update ----
    #pragma unroll
    for (int j = 0; j < 2; ++j) {
      v8f acc = l0 ? gemm_job<true, true >(Abase, WcBase[j])
                   : gemm_job<true, false>(Abase, WcBase[j]);
      int n = nC[j];
      float bias = bcv[j];
      #pragma unroll
      for (int i = 0; i < 8; ++i) {
        int m = mrow0 + i;
        float c  = tanh_f(acc[i] + bias);
        float u  = U[m * 256 + n];
        float hn = u * H[m * 256 + n] + (1.0f - u) * c;
        H[m * 256 + n] = hn;
        A[m * 768 + 256 + n] = (__bf16)hn;            // h for next step's gates
        ysOut[(long)t * BATCH * HID + m * 256 + n] = (__bf16)hn;
        if (L == 2 && t == T_STEPS - 1) hlast[m * 256 + n] = hn;
      }
    }
    __threadfence();
    __syncthreads();
    if (tid == 0)
      __hip_atomic_store(&flags[L], t + 1, __ATOMIC_RELEASE, __HIP_MEMORY_SCOPE_AGENT);
  }
}

// ---------------------------------------------------------------------------
// Final projection: logits[64][10] = h_last @ Wout + bout
// ---------------------------------------------------------------------------
__global__ void out_proj(const float* __restrict__ hlast, const float* __restrict__ Wout,
                         const float* __restrict__ bout, float* __restrict__ out) {
  int i = blockIdx.x * blockDim.x + threadIdx.x;
  if (i >= 64 * 10) return;
  int b = i / 10, o = i - b * 10;
  float s = bout[o];
  for (int k = 0; k < 256; ++k) s += hlast[b * 256 + k] * Wout[k * 10 + o];
  out[i] = s;
}

extern "C" void kernel_launch(void* const* d_in, const int* in_sizes, int n_in,
                              void* d_out, int out_size, void* d_ws, size_t ws_size,
                              hipStream_t stream) {
  (void)in_sizes; (void)n_in; (void)out_size; (void)ws_size;
  const float* x    = (const float*)d_in[0];
  const float* Wg0  = (const float*)d_in[1];
  const float* bg0  = (const float*)d_in[2];
  const float* Wc0  = (const float*)d_in[3];
  const float* bc0  = (const float*)d_in[4];
  const float* Wg1  = (const float*)d_in[5];
  const float* bg1  = (const float*)d_in[6];
  const float* Wc1  = (const float*)d_in[7];
  const float* bc1  = (const float*)d_in[8];
  const float* Wg2  = (const float*)d_in[9];
  const float* bg2  = (const float*)d_in[10];
  const float* Wc2  = (const float*)d_in[11];
  const float* bc2  = (const float*)d_in[12];
  const float* Wout = (const float*)d_in[13];
  const float* bout = (const float*)d_in[14];
  char* ws = (char*)d_ws;

  prep_weights<<<(1179648 + 255) / 256, 256, 0, stream>>>(Wg0, Wc0, Wg1, Wc1, Wg2, Wc2, ws);

  const int shmem = 64 * 768 * 2 + 2 * 64 * 256 * 4;   // 229376 B (< 320KB WGP LDS)
  (void)hipFuncSetAttribute((const void*)gru_stack,
                            hipFuncAttributeMaxDynamicSharedMemorySize, shmem);
  gru_stack<<<3, 1024, shmem, stream>>>(x, bg0, bc0, bg1, bc1, bg2, bc2, ws);

  out_proj<<<3, 256, 0, stream>>>((const float*)(ws + OFF_HLAST), Wout, bout, (float*)d_out);
}
